// SpikingMinGRUBlock_23776938951037
// MI455X (gfx1250) — matrix operand order repaired and checked by reading, not compiled, not run on hardware
//
#include <hip/hip_runtime.h>
#include <hip/hip_bf16.h>
#include <math.h>

// ---------------- problem constants ----------------
#define Bv       8
#define Lv       2048
#define DIM      1024
#define HEADS    8
#define HEAD_DIM 128
#define INNER    192
#define HI       (HEADS * INNER)      // 1536
#define FI       2730
#define FIP      2752                 // FI padded to multiple of 64 (tiles + 16B align)
#define BL       (Bv * Lv)            // 16384

typedef __attribute__((ext_vector_type(16))) __bf16 v16bf;
typedef __attribute__((ext_vector_type(8)))  __bf16 v8bf;
typedef __attribute__((ext_vector_type(8)))  float  v8f;

union U16BF { v16bf v; v8bf h[2]; };

// A fragment: 16x32 bf16, row-major A with leading dim lda.
// ISA layout: lane holds row M=lane&15; lanes 0-15: K in {0..7,16..23},
// lanes 16-31: K in {8..15,24..31}.
__device__ __forceinline__ v16bf load_a_frag(const __bf16* A, int lda,
                                             int m0, int k0, int lane) {
    int m   = m0 + (lane & 15);
    int klo = (lane < 16) ? 0 : 8;
    const __bf16* p = A + (size_t)m * lda + k0 + klo;
    U16BF u;
    u.h[0] = *(const v8bf*)(p);
    u.h[1] = *(const v8bf*)(p + 16);
    return u.v;
}

// B fragment: 32x16 (KxN) from W stored row-major [N, K] (ldb = row stride).
// ISA layout: lane holds column N=lane&15; lanes 0-15: K in {0..7,16..23},
// lanes 16-31: K in {8..15,24..31}. No bounds check: N is pre-padded.
__device__ __forceinline__ v16bf load_b_frag(const __bf16* W, int ldb,
                                             int n0, int k0, int lane) {
    int n    = n0 + (lane & 15);
    int koff = (lane < 16) ? 0 : 8;
    const __bf16* p = W + (size_t)n * ldb + k0 + koff;
    U16BF u;
    u.h[0] = *(const v8bf*)(p);
    u.h[1] = *(const v8bf*)(p + 16);
    return u.v;
}

// C/D fragment store: VGPR r, lane -> (M = m0 + r + (lane<16?0:8), N = n0 + lane&15)
__device__ __forceinline__ void store_c_frag(float* C, int ldc, int m0, int n0,
                                             v8f c, int lane) {
    int n  = n0 + (lane & 15);
    int mr = m0 + ((lane < 16) ? 0 : 8);
    #pragma unroll
    for (int r = 0; r < 8; ++r)
        C[(size_t)(mr + r) * ldc + n] = c[r];
}

// ---------------- NT GEMM: C[M,Npad] = A[M,K] * W[Npad,K]^T, bf16 in / f32 out
// grid: (ceil(Npad/128), M/128, Z). 256-thread block = 8 waves arranged 4(M)x2(N);
// each wave computes a 32x64 tile: 8 bf16 WMMAs per K=32 step, zero guards in
// the hot loop (N pre-padded to a multiple of 64, K to a multiple of 32).
__global__ void __launch_bounds__(256)
gemm_nt_bf16(const __bf16* __restrict__ A, int lda, long long sAz,
             const __bf16* __restrict__ W, int ldb, long long sBz,
             float* __restrict__ C, int ldc, long long sCz,
             int Npad, int K) {
    A += (size_t)blockIdx.z * sAz;
    W += (size_t)blockIdx.z * sBz;
    C += (size_t)blockIdx.z * sCz;

    const int wave  = threadIdx.x >> 5;
    const int lane  = threadIdx.x & 31;
    const int waveM = wave >> 1;            // 0..3
    const int waveN = wave & 1;             // 0..1
    const int m0 = blockIdx.y * 128 + waveM * 32;
    const int n0 = blockIdx.x * 128 + waveN * 64;
    if (n0 >= Npad) return;                 // wave-uniform skip of padded tiles

    v8f zero = {};
    v8f acc[2][4];
    #pragma unroll
    for (int i = 0; i < 2; ++i)
        #pragma unroll
        for (int j = 0; j < 4; ++j) acc[i][j] = zero;

    for (int k0 = 0; k0 < K; k0 += 32) {
        // speculative prefetch of the next K tile (global_prefetch_b8)
        __builtin_prefetch(A + (size_t)(m0 + (lane & 15)) * lda + k0 + 32, 0, 1);
        __builtin_prefetch(W + (size_t)(n0 + (lane & 15)) * ldb + k0 + 32, 0, 1);

        v16bf a0 = load_a_frag(A, lda, m0,      k0, lane);
        v16bf a1 = load_a_frag(A, lda, m0 + 16, k0, lane);
        v16bf b[4];
        #pragma unroll
        for (int j = 0; j < 4; ++j)
            b[j] = load_b_frag(W, ldb, n0 + 16 * j, k0, lane);

        #pragma unroll
        for (int j = 0; j < 4; ++j) {
            acc[0][j] = __builtin_amdgcn_wmma_f32_16x16x32_bf16(false, a0, false, b[j], (short)0, acc[0][j], false, false);
            acc[1][j] = __builtin_amdgcn_wmma_f32_16x16x32_bf16(false, a1, false, b[j], (short)0, acc[1][j], false, false);
        }
    }

    #pragma unroll
    for (int i = 0; i < 2; ++i)
        #pragma unroll
        for (int j = 0; j < 4; ++j)
            store_c_frag(C, ldc, m0 + 16 * i, n0 + 16 * j, acc[i][j], lane);
}

// ------------- f32 -> bf16 weight convert with row AND column zero-pad -------
__global__ void __launch_bounds__(256)
convert_pad_kernel(const float* __restrict__ src, __bf16* __restrict__ dst,
                   int rows, int cols, int rpad, int ldd) {
    long long idx = (long long)blockIdx.x * blockDim.x + threadIdx.x;
    long long total = (long long)rpad * ldd;
    if (idx >= total) return;
    int r = (int)(idx / ldd), c = (int)(idx % ldd);
    dst[idx] = (r < rows && c < cols) ? (__bf16)src[(size_t)r * cols + c] : (__bf16)0.0f;
}

// ---------------- RMSNorm: f32 [rows, D] -> bf16, optional (B,L)->(L,B) ------
__global__ void __launch_bounds__(256)
rmsnorm_bf16_kernel(const float* __restrict__ x, const float* __restrict__ gamma,
                    __bf16* __restrict__ out, int D, int transposeLB) {
    const int row = blockIdx.x;                 // m = b*L + l
    const float* xr = x + (size_t)row * D;
    __shared__ float red[8];
    __shared__ float stotal;

    float ss = 0.0f;
    for (int d = threadIdx.x; d < D; d += 256) { float v = xr[d]; ss += v * v; }
    #pragma unroll
    for (int off = 16; off > 0; off >>= 1) ss += __shfl_down(ss, off, 32);
    if ((threadIdx.x & 31) == 0) red[threadIdx.x >> 5] = ss;
    __syncthreads();
    if (threadIdx.x == 0) {
        float t = 0.0f;
        #pragma unroll
        for (int w = 0; w < 8; ++w) t += red[w];
        stotal = t;
    }
    __syncthreads();

    float scale = sqrtf((float)D) / fmaxf(sqrtf(stotal), 1e-12f);
    int orow = row;
    if (transposeLB) { int b = row / Lv, l = row % Lv; orow = l * Bv + b; }
    __bf16* orp = out + (size_t)orow * D;
    for (int d = threadIdx.x; d < D; d += 256)
        orp[d] = (__bf16)(xr[d] * scale * (gamma[d] + 1.0f));
}

// ---------------- sequential LIF/minGRU scan (elementwise only) --------------
// gc/cc layout: [L, B, H*INNER] (row m = l*B+b from the GEMM). One thread per
// (b, h, i) state, loops over L. Writes h_t as bf16 in the same layout.
__global__ void __launch_bounds__(256)
scan_kernel(const float* __restrict__ gc, const float* __restrict__ cc,
            const float* __restrict__ g_beta_raw, const float* __restrict__ g_theta,
            const float* __restrict__ c_beta_raw, const float* __restrict__ c_theta,
            __bf16* __restrict__ hbuf) {
    const int tid = blockIdx.x * blockDim.x + threadIdx.x;   // b*HI + hi
    if (tid >= Bv * HI) return;
    const int hi = tid % HI;

    const float bg = 1.0f / (1.0f + __expf(-g_beta_raw[hi]));
    const float bc = 1.0f / (1.0f + __expf(-c_beta_raw[hi]));
    const float tg = g_theta[hi];
    const float tc = c_theta[hi];

    float h = 0.0f, gm = 0.0f, cm = 0.0f;
    const int S = Bv * HI;                                   // 12288
    for (int l = 0; l < Lv; ++l) {
        const size_t off = (size_t)l * S + tid;
        gm = bg * gm + gc[off];
        float gs = fminf(fmaxf(rintf(gm), -1.0f), 1.0f);     // RNE round == jnp.round
        gm -= gs * tg;
        float z = (gs + 1.0f) * 0.5f;
        cm = bc * cm + cc[off];
        float cs = fminf(fmaxf(rintf(cm), -1.0f), 1.0f);
        cm -= cs * tc;
        h = (1.0f - z) * h + z * cs;
        hbuf[off] = (__bf16)h;
    }
}

// ---------------- residual: y[b,l,d] = x[b,l,d] + gru[l,b,d] -----------------
__global__ void __launch_bounds__(256)
residual_t_kernel(const float* __restrict__ x, const float* __restrict__ gru,
                  float* __restrict__ y) {
    long long e = (long long)blockIdx.x * blockDim.x + threadIdx.x;
    if (e >= (long long)BL * DIM) return;
    int d = (int)(e % DIM);
    long long t = e / DIM;
    int l = (int)(t % Lv), b = (int)(t / Lv);
    y[e] = x[e] + gru[((size_t)l * Bv + b) * DIM + d];
}

// -------- SwiGLU gate: act = silu(g)*u, all buffers use ld = FIP -------------
// Padded columns of g/u are exact zeros (zero weight rows), so no branch needed.
__global__ void __launch_bounds__(256)
silu_mul_kernel(const float* __restrict__ g, const float* __restrict__ u,
                __bf16* __restrict__ act) {
    long long e = (long long)blockIdx.x * blockDim.x + threadIdx.x;
    if (e >= (long long)BL * FIP) return;
    float gv = g[e];
    float uv = u[e];
    act[e] = (__bf16)(gv / (1.0f + __expf(-gv)) * uv);
}

// ---------------- final residual: out = y + ff -------------------------------
__global__ void __launch_bounds__(256)
add_kernel(const float* __restrict__ y, const float* __restrict__ ff,
           float* __restrict__ out) {
    long long e = (long long)blockIdx.x * blockDim.x + threadIdx.x;
    if (e >= (long long)BL * DIM) return;
    out[e] = y[e] + ff[e];
}

// ============================================================================
extern "C" void kernel_launch(void* const* d_in, const int* in_sizes, int n_in,
                              void* d_out, int out_size, void* d_ws, size_t ws_size,
                              hipStream_t stream) {
    const float* x        = (const float*)d_in[0];
    const float* in_gamma = (const float*)d_in[1];
    const float* gate_W   = (const float*)d_in[2];
    const float* cand_W   = (const float*)d_in[3];
    const float* out_W    = (const float*)d_in[4];
    const float* g_braw   = (const float*)d_in[5];
    const float* g_theta  = (const float*)d_in[6];
    const float* c_braw   = (const float*)d_in[7];
    const float* c_theta  = (const float*)d_in[8];
    const float* ff_gamma = (const float*)d_in[9];
    const float* ffg_W    = (const float*)d_in[10];
    const float* ffu_W    = (const float*)d_in[11];
    const float* ffd_W    = (const float*)d_in[12];
    float* out = (float*)d_out;

    // ---- workspace layout (256B aligned) ----
    char* ws = (char*)d_ws;
    size_t off = 0;
    auto alloc = [&](size_t bytes) { void* p = ws + off; off += (bytes + 255) & ~(size_t)255; return p; };
    __bf16* xnt    = (__bf16*)alloc((size_t)BL * DIM * 2);       // [L,B,D] bf16
    __bf16* gateWb = (__bf16*)alloc((size_t)HI * HEAD_DIM * 2);
    __bf16* candWb = (__bf16*)alloc((size_t)HI * HEAD_DIM * 2);
    __bf16* outWb  = (__bf16*)alloc((size_t)HEADS * HEAD_DIM * INNER * 2);
    __bf16* ffgWb  = (__bf16*)alloc((size_t)FIP * DIM * 2);      // rows padded -> 0
    __bf16* ffuWb  = (__bf16*)alloc((size_t)FIP * DIM * 2);
    __bf16* ffdWb  = (__bf16*)alloc((size_t)DIM * FIP * 2);      // cols padded -> 0
    float*  gc     = (float*) alloc((size_t)BL * HI * 4);
    float*  cc     = (float*) alloc((size_t)BL * HI * 4);
    __bf16* hbuf   = (__bf16*)alloc((size_t)BL * HI * 2);
    float*  gru    = (float*) alloc((size_t)BL * DIM * 4);       // [L,B,D]
    float*  ybuf   = (float*) alloc((size_t)BL * DIM * 4);       // [B,L,D]
    __bf16* fnb    = (__bf16*)alloc((size_t)BL * DIM * 2);
    float*  gbuf   = (float*) alloc((size_t)BL * FIP * 4);       // ld = FIP
    float*  ubuf   = (float*) alloc((size_t)BL * FIP * 4);
    __bf16* actb   = (__bf16*)alloc((size_t)BL * FIP * 2);
    float*  ffbuf  = (float*) alloc((size_t)BL * DIM * 4);
    (void)ws_size; (void)in_sizes; (void)n_in; (void)out_size;

    auto blocks1d = [](long long n) { return (unsigned)((n + 255) / 256); };

    // ---- 0) weight conversion to bf16 with zero padding ----
    convert_pad_kernel<<<blocks1d((long long)HI*HEAD_DIM), 256, 0, stream>>>(gate_W, gateWb, HI, HEAD_DIM, HI, HEAD_DIM);
    convert_pad_kernel<<<blocks1d((long long)HI*HEAD_DIM), 256, 0, stream>>>(cand_W, candWb, HI, HEAD_DIM, HI, HEAD_DIM);
    convert_pad_kernel<<<blocks1d((long long)HEADS*HEAD_DIM*INNER), 256, 0, stream>>>(out_W, outWb, HEADS*HEAD_DIM, INNER, HEADS*HEAD_DIM, INNER);
    convert_pad_kernel<<<blocks1d((long long)FIP*DIM), 256, 0, stream>>>(ffg_W, ffgWb, FI, DIM, FIP, DIM);
    convert_pad_kernel<<<blocks1d((long long)FIP*DIM), 256, 0, stream>>>(ffu_W, ffuWb, FI, DIM, FIP, DIM);
    convert_pad_kernel<<<blocks1d((long long)DIM*FIP), 256, 0, stream>>>(ffd_W, ffdWb, DIM, FI, DIM, FIP);

    // ---- 1) RMSNorm(x) -> bf16, transposed to [L,B,D] for the scan layout ----
    rmsnorm_bf16_kernel<<<BL, 256, 0, stream>>>(x, in_gamma, xnt, DIM, 1);

    // ---- 2) hoisted gate/candidate projections as batched GEMMs over heads ----
    {
        dim3 grid(2, BL / 128, HEADS);   // covers N up to 256; Npad=192 skips rest
        gemm_nt_bf16<<<grid, 256, 0, stream>>>(xnt, DIM, HEAD_DIM,
                                               gateWb, HEAD_DIM, (long long)INNER*HEAD_DIM,
                                               gc, HI, INNER, INNER, HEAD_DIM);
        gemm_nt_bf16<<<grid, 256, 0, stream>>>(xnt, DIM, HEAD_DIM,
                                               candWb, HEAD_DIM, (long long)INNER*HEAD_DIM,
                                               cc, HI, INNER, INNER, HEAD_DIM);
    }

    // ---- 3) sequential elementwise recurrence (only true dependency chain) ----
    scan_kernel<<<blocks1d(Bv * HI), 256, 0, stream>>>(gc, cc, g_braw, g_theta,
                                                       c_braw, c_theta, hbuf);

    // ---- 4) output projection as batched GEMM: gru[L,B,D] = h * out_W^T ----
    {
        dim3 grid(1, BL / 128, HEADS);
        gemm_nt_bf16<<<grid, 256, 0, stream>>>(hbuf, HI, INNER,
                                               outWb, INNER, (long long)HEAD_DIM*INNER,
                                               gru, DIM, HEAD_DIM, HEAD_DIM, INNER);
    }

    // ---- 5) residual (with [L,B]->[B,L] transpose) + RMSNorm -> bf16 ----
    residual_t_kernel<<<blocks1d((long long)BL * DIM), 256, 0, stream>>>(x, gru, ybuf);
    rmsnorm_bf16_kernel<<<BL, 256, 0, stream>>>(ybuf, ff_gamma, fnb, DIM, 0);

    // ---- 6) FFN: g = fn*Wg^T, u = fn*Wu^T, act = silu(g)*u, ff = act*Wd^T ----
    {
        dim3 grid((FIP + 127) / 128, BL / 128, 1);   // 22 x 128
        gemm_nt_bf16<<<grid, 256, 0, stream>>>(fnb, DIM, 0, ffgWb, DIM, 0,
                                               gbuf, FIP, 0, FIP, DIM);
        gemm_nt_bf16<<<grid, 256, 0, stream>>>(fnb, DIM, 0, ffuWb, DIM, 0,
                                               ubuf, FIP, 0, FIP, DIM);
    }
    silu_mul_kernel<<<blocks1d((long long)BL * FIP), 256, 0, stream>>>(gbuf, ubuf, actb);
    {
        dim3 grid(DIM / 128, BL / 128, 1);
        gemm_nt_bf16<<<grid, 256, 0, stream>>>(actb, FIP, 0, ffdWb, FIP, 0,
                                               ffbuf, DIM, 0, DIM, FIP);
    }

    // ---- 7) final residual -> out ----
    add_kernel<<<blocks1d((long long)BL * DIM), 256, 0, stream>>>(ybuf, ffbuf, out);
}
